// GNODecoder_8684423873090
// MI455X (gfx1250) — matrix operand order repaired and checked by reading, not compiled, not run on hardware
//
#include <hip/hip_runtime.h>
#include <hip/hip_bf16.h>

typedef __attribute__((ext_vector_type(16))) _Float16 v16h;
typedef __attribute__((ext_vector_type(8)))  float    v8f;

// Branchless GELU (tanh form), one v_exp_f32 + one v_rcp_f32:
// gelu(x) ~= x - x * rcp(exp2(x*(k1 + k2*x^2)) + 1)
// k1 = 2*sqrt(2/pi)*log2(e), k2 = k1*0.044715
__device__ __forceinline__ float gelu_fast(float x) {
    float x2 = x * x;
    float t  = x * (2.30218645f + 0.10294227f * x2);
    float e  = __builtin_amdgcn_exp2f(t);
    float d  = __builtin_amdgcn_rcpf(e + 1.0f);
    return x - x * d;      // x*(1 - sigmoid-like) -> exact limits at +/-inf
}

// A-fragment (16x32 f16, M x K) per ISA 7.12.2: lanes 0-15 are rows M=0..15,
// VGPR 0-3 hold K = 8*g + {0..7}, VGPR 4-7 hold K = 16 + 8*g + {0..7},
// g = lane/16. LDS layout: row-major [M][ldk], K contiguous -> 32-bit reads.
__device__ __forceinline__ v16h load_a_frag(const _Float16* base, int row0, int ldk,
                                            int k0, int lane) {
    const int g = (lane >> 4) & 1;
    const int m = lane & 15;
    union { v16h v; unsigned u[8]; } r;
    const unsigned* p = (const unsigned*)(base + (row0 + m) * ldk + k0);
#pragma unroll
    for (int i = 0; i < 8; ++i) {
        int k = (i < 4) ? (8 * g + 2 * i) : (16 + 8 * g + 2 * (i - 4));
        r.u[i] = p[k >> 1];
    }
    return r.v;
}

// B-fragment (32x16 f16, K x N): same K order as A: VGPR 0-3 hold K = 8*g + {0..7},
// VGPR 4-7 hold K = 16 + 8*g + {0..7}; N = lane&15. Weights staged as [N][ldk] in LDS.
__device__ __forceinline__ v16h load_b_frag(const _Float16* base, int n0, int ldk,
                                            int k0, int lane) {
    const int g = (lane >> 4) & 1;
    const int n = lane & 15;
    union { v16h v; unsigned u[8]; } r;
    const unsigned* p = (const unsigned*)(base + (n0 + n) * ldk + k0);
#pragma unroll
    for (int vv = 0; vv < 8; ++vv) {
        r.u[vv] = p[((vv < 4) ? (8 * g + 2 * vv) : (16 + 8 * g + 2 * (vv - 4))) >> 1];
    }
    return r.v;
}

#define WMMA_F16(A, B, C) \
    __builtin_amdgcn_wmma_f32_16x16x32_f16(false, (A), false, (B), (short)0, (C), false, false)

// ---------------------------------------------------------------------------
// Kernel 1: per-edge kernel-MLP (6->64->64->64, GELU between) fused with
// kernel*rndata[src] and the 16-edge mean per query.
// Block: 256 threads = 8 waves; weights staged once, then ITER1=5 tiles of
// 16 queries (256 edges each). EF/H/src rows are wave-private (wave w owns
// rows 32w..32w+31 at every layer), so the iteration loop needs NO barriers.
// ---------------------------------------------------------------------------
#define ITER1 5

__global__ __launch_bounds__(256) void gno_edge_mlp_reduce(
    const float* __restrict__ rndata,   // [NL,64]
    const float* __restrict__ qpos,     // [NQ,3]
    const float* __restrict__ lpos,     // [NL,3]
    const int*   __restrict__ esrc,     // [E] = edge_index row 1
    const float* __restrict__ kW0, const float* __restrict__ kb0,
    const float* __restrict__ kW1, const float* __restrict__ kb1,
    const float* __restrict__ kW2, const float* __restrict__ kb2,
    float* __restrict__ decoded,        // [NQ,64]
    int NQ, int Etot)
{
    extern __shared__ char smem_raw[];
    _Float16* wt0 = (_Float16*)(smem_raw);           // [64][32] kW0^T, K-padded 6->32
    _Float16* wt1 = (_Float16*)(smem_raw + 4096);    // [64][64] kW1^T
    _Float16* wt2 = (_Float16*)(smem_raw + 12288);   // [64][64] kW2^T
    _Float16* EF  = (_Float16*)(smem_raw + 20480);   // [256][32] edge features
    _Float16* H   = (_Float16*)(smem_raw + 36864);   // [256][64] hidden (in-place L1->L2)
    float* b0s = (float*)(smem_raw + 69632);         // [64]
    float* b1s = (float*)(smem_raw + 69888);         // [64]
    float* b2s = (float*)(smem_raw + 70144);         // [64]
    int*   srcl = (int*)(smem_raw + 70400);          // [256]

    const int tid = threadIdx.x;

    // --- stage weights (transposed, f16) + biases: once per block ---
    for (int i = tid; i < 64 * 32; i += 256) {
        int n = i >> 5, k = i & 31;
        wt0[i] = (k < 6) ? (_Float16)kW0[k * 64 + n] : (_Float16)0.0f;
    }
    for (int i = tid; i < 64 * 64; i += 256) {
        int n = i >> 6, k = i & 63;
        wt1[i] = (_Float16)kW1[k * 64 + n];
        wt2[i] = (_Float16)kW2[k * 64 + n];
    }
    if (tid < 64) { b0s[tid] = kb0[tid]; b1s[tid] = kb1[tid]; b2s[tid] = kb2[tid]; }
    __syncthreads();   // only barrier needed: weights are read-only afterwards

    const int lane = tid & 31;
    const int wv   = tid >> 5;
    const int g    = lane >> 4;
    const int nlo  = lane & 15;

    for (int it = 0; it < ITER1; ++it) {
        const int qbase = (blockIdx.x * ITER1 + it) * 16;

        // --- stage edge features (wave-private rows): [lpos[src], qpos[dst], 0..] ---
        {
            int e = qbase * 16 + tid;      // one edge per thread
            _Float16* row = EF + tid * 32;
            if (e < Etot) {
                int s = esrc[e];
                srcl[tid] = s;
                int q = e >> 4;            // dst = repeat(arange(NQ),16)
                row[0] = (_Float16)lpos[s * 3 + 0];
                row[1] = (_Float16)lpos[s * 3 + 1];
                row[2] = (_Float16)lpos[s * 3 + 2];
                row[3] = (_Float16)qpos[q * 3 + 0];
                row[4] = (_Float16)qpos[q * 3 + 1];
                row[5] = (_Float16)qpos[q * 3 + 2];
            } else {
                srcl[tid] = 0;
#pragma unroll
                for (int k = 0; k < 6; ++k) row[k] = (_Float16)0.0f;
            }
#pragma unroll
            for (int k = 6; k < 32; ++k) row[k] = (_Float16)0.0f;
        }

        // --- layer 0: [256x32] x [32x64] -> GELU -> H ---
#pragma unroll
        for (int mt = 0; mt < 2; ++mt) {
            int row0 = (2 * wv + mt) * 16;
            v16h a = load_a_frag(EF, row0, 32, 0, lane);
#pragma unroll
            for (int nb = 0; nb < 4; ++nb) {
                v8f acc = {};
                acc = WMMA_F16(a, load_b_frag(wt0, nb * 16, 32, 0, lane), acc);
                int col = nb * 16 + nlo;
                float bias = b0s[col];
#pragma unroll
                for (int r = 0; r < 8; ++r)   // D: (row=r+8g, col=lane&15)
                    H[(row0 + r + 8 * g) * 64 + col] = (_Float16)gelu_fast(acc[r] + bias);
            }
        }

        // --- layer 1: [256x64] x [64x64] -> GELU -> H (in place, row-disjoint tiles) ---
#pragma unroll
        for (int mt = 0; mt < 2; ++mt) {
            int row0 = (2 * wv + mt) * 16;
            v16h a0 = load_a_frag(H, row0, 64, 0, lane);
            v16h a1 = load_a_frag(H, row0, 64, 32, lane);
#pragma unroll
            for (int nb = 0; nb < 4; ++nb) {
                v8f acc = {};
                acc = WMMA_F16(a0, load_b_frag(wt1, nb * 16, 64, 0, lane), acc);
                acc = WMMA_F16(a1, load_b_frag(wt1, nb * 16, 64, 32, lane), acc);
                int col = nb * 16 + nlo;
                float bias = b1s[col];
#pragma unroll
                for (int r = 0; r < 8; ++r)
                    H[(row0 + r + 8 * g) * 64 + col] = (_Float16)gelu_fast(acc[r] + bias);
            }
        }

        // --- layer 2 + kernel*rndata[src] + mean over 16 edges of each query ---
#pragma unroll
        for (int mt = 0; mt < 2; ++mt) {
            int t = 2 * wv + mt;
            int row0 = t * 16;
            int q = qbase + t;
            v16h a0 = load_a_frag(H, row0, 64, 0, lane);
            v16h a1 = load_a_frag(H, row0, 64, 32, lane);
#pragma unroll
            for (int nb = 0; nb < 4; ++nb) {
                v8f acc = {};
                acc = WMMA_F16(a0, load_b_frag(wt2, nb * 16, 64, 0, lane), acc);
                acc = WMMA_F16(a1, load_b_frag(wt2, nb * 16, 64, 32, lane), acc);
                int col = nb * 16 + nlo;
                float bias = b2s[col];
                float s = 0.0f;
#pragma unroll
                for (int r = 0; r < 8; ++r) {
                    int er = row0 + r + 8 * g;        // edge row inside tile set
                    int si = srcl[er];
                    s += (acc[r] + bias) * rndata[si * 64 + col]; // L2-hot gather (4MB)
                }
                s += __shfl_xor(s, 16, 32);           // add other half's 8 rows
                if (lane < 16 && q < NQ)
                    decoded[q * 64 + col] = s * 0.0625f;  // /16 (fixed degree)
            }
        }
    }
}

// ---------------------------------------------------------------------------
// Kernel 2: out = gelu(decoded @ pW1 + pb1) @ pW2 + pb2.
// Block: 128 threads = 4 waves; weights staged once, then each wave loops
// ITER2=4 private 16-query tiles (256 queries/block). dec/hid are wave-private
// so the loop needs no barriers.
// ---------------------------------------------------------------------------
#define ITER2 4

__global__ __launch_bounds__(128) void gno_project(
    const float* __restrict__ decoded,  // [NQ,64]
    const float* __restrict__ pW1, const float* __restrict__ pb1,
    const float* __restrict__ pW2, const float* __restrict__ pb2,
    float* __restrict__ out, int NQ)
{
    extern __shared__ char smem_raw[];
    _Float16* w1t  = (_Float16*)(smem_raw);            // [256][64]  pW1^T
    _Float16* w2t  = (_Float16*)(smem_raw + 32768);    // [16][256]  pW2^T padded 4->16
    _Float16* decs = (_Float16*)(smem_raw + 40960);    // 4 x [16][64]
    _Float16* hids = (_Float16*)(smem_raw + 49152);    // 4 x [16][256]
    float* pb1s = (float*)(smem_raw + 81920);          // [256]
    float* pb2s = (float*)(smem_raw + 82944);          // [4]

    const int tid  = threadIdx.x;
    const int lane = tid & 31;
    const int wv   = tid >> 5;

    for (int i = tid; i < 256 * 64; i += 128) {
        int n = i >> 6, k = i & 63;
        w1t[i] = (_Float16)pW1[k * 256 + n];
    }
    for (int i = tid; i < 16 * 256; i += 128) {
        int n = i >> 8, k = i & 255;
        w2t[i] = (n < 4) ? (_Float16)pW2[k * 4 + n] : (_Float16)0.0f;
    }
    for (int i = tid; i < 256; i += 128) pb1s[i] = pb1[i];
    if (tid < 4) pb2s[tid] = pb2[tid];
    __syncthreads();

    const int g   = lane >> 4;
    const int nlo = lane & 15;
    _Float16* dec = decs + wv * (16 * 64);
    _Float16* hid = hids + wv * (16 * 256);

    for (int it = 0; it < ITER2; ++it) {
        const int q0 = (blockIdx.x * ITER2 + it) * 64 + wv * 16;

        // stage one 16x64 decoded tile (wave-private), f32 -> f16
        for (int i = lane; i < 16 * 64; i += 32) {
            int rr = i >> 6, cc = i & 63;
            int q = q0 + rr;
            dec[i] = (q < NQ) ? (_Float16)decoded[q * 64 + cc] : (_Float16)0.0f;
        }

        // hidden = gelu(dec @ pW1 + pb1): 16 N-tiles x (K=64 -> 2 WMMA)
        v16h a0 = load_a_frag(dec, 0, 64, 0, lane);
        v16h a1 = load_a_frag(dec, 0, 64, 32, lane);
#pragma unroll
        for (int nb = 0; nb < 16; ++nb) {
            v8f acc = {};
            acc = WMMA_F16(a0, load_b_frag(w1t, nb * 16, 64, 0, lane), acc);
            acc = WMMA_F16(a1, load_b_frag(w1t, nb * 16, 64, 32, lane), acc);
            int col = nb * 16 + nlo;
            float bias = pb1s[col];
#pragma unroll
            for (int r = 0; r < 8; ++r)
                hid[(r + 8 * g) * 256 + col] = (_Float16)gelu_fast(acc[r] + bias);
        }

        // out = hidden @ pW2 + pb2 : one D-tile, K=256 -> 8 WMMA
        v8f acc = {};
#pragma unroll
        for (int kc = 0; kc < 8; ++kc) {
            v16h a = load_a_frag(hid, 0, 256, kc * 32, lane);
            v16h b = load_b_frag(w2t, 0, 256, kc * 32, lane);
            acc = WMMA_F16(a, b, acc);
        }
#pragma unroll
        for (int r = 0; r < 8; ++r) {
            int q = q0 + r + 8 * g;
            if (q < NQ && nlo < 4) out[q * 4 + nlo] = acc[r] + pb2s[nlo];
        }
    }
}

extern "C" void kernel_launch(void* const* d_in, const int* in_sizes, int n_in,
                              void* d_out, int out_size, void* d_ws, size_t ws_size,
                              hipStream_t stream) {
    const float* rndata = (const float*)d_in[0];
    const float* qpos   = (const float*)d_in[1];
    const float* lpos   = (const float*)d_in[3];
    const int*   eidx   = (const int*)d_in[5];
    const float* kW0 = (const float*)d_in[6];
    const float* kb0 = (const float*)d_in[7];
    const float* kW1 = (const float*)d_in[8];
    const float* kb1 = (const float*)d_in[9];
    const float* kW2 = (const float*)d_in[10];
    const float* kb2 = (const float*)d_in[11];
    const float* pW1 = (const float*)d_in[12];
    const float* pb1 = (const float*)d_in[13];
    const float* pW2 = (const float*)d_in[14];
    const float* pb2 = (const float*)d_in[15];
    float* out = (float*)d_out;
    float* decoded = (float*)d_ws;          // [NQ,64] f32 scratch (25.6 MB, L2-resident)

    const int NQ = in_sizes[1] / 3;         // 100000
    const int E  = in_sizes[5] / 2;         // 1.6M
    const int* esrc = eidx + E;             // edge_index row 1

    const size_t lds1 = 71424;
    const size_t lds2 = 82960;
    (void)hipFuncSetAttribute((const void*)gno_edge_mlp_reduce,
                              hipFuncAttributeMaxDynamicSharedMemorySize, (int)lds1);
    (void)hipFuncSetAttribute((const void*)gno_project,
                              hipFuncAttributeMaxDynamicSharedMemorySize, (int)lds2);

    const int blocks1 = (NQ + 16 * ITER1 - 1) / (16 * ITER1);   // 1250
    const int blocks2 = (NQ + 64 * ITER2 - 1) / (64 * ITER2);   // 391
    gno_edge_mlp_reduce<<<blocks1, 256, lds1, stream>>>(
        rndata, qpos, lpos, esrc, kW0, kb0, kW1, kb1, kW2, kb2, decoded, NQ, E);
    gno_project<<<blocks2, 128, lds2, stream>>>(
        decoded, pW1, pb1, pW2, pb2, out, NQ);
}
